// Position_AM_Module_3D_52596169507374
// MI455X (gfx1250) — matrix operand order repaired and checked by reading, not compiled, not run on hardware
//
#include <hip/hip_runtime.h>
#include <hip/hip_bf16.h>

#ifndef __has_builtin
#define __has_builtin(x) 0
#endif

#if defined(__gfx1250__) && __has_builtin(__builtin_amdgcn_global_load_async_to_lds_b128) && __has_builtin(__builtin_amdgcn_s_wait_asynccnt)
#define HAVE_ASYNC 1
#else
#define HAVE_ASYNC 0
#endif

typedef __attribute__((ext_vector_type(16))) _Float16 v16h;
typedef __attribute__((ext_vector_type(8)))  _Float16 v8h;
typedef __attribute__((ext_vector_type(8)))  float    v8f;
typedef __attribute__((ext_vector_type(4)))  int      v4i;

#if HAVE_ASYNC
typedef __attribute__((address_space(1))) v4i gv4i_t;   // global 16B chunk
typedef __attribute__((address_space(3))) v4i sv4i_t;   // LDS    16B chunk
#endif

#define BB 2
#define CC 256
#define DD 64
#define NN 8192

// ---------- helpers ----------

__device__ __forceinline__ v16h ld16_split(const _Float16* p0, const _Float16* p1) {
    v8h a = *(const v8h*)p0;
    v8h b = *(const v8h*)p1;
    v16h r;
#pragma unroll
    for (int i = 0; i < 8; ++i) { r[i] = a[i]; r[i + 8] = b[i]; }
    return r;
}

__device__ __forceinline__ v16h ld16_contig(const _Float16* p) {
    return ld16_split(p, p + 8);
}

__device__ __forceinline__ v8f wmma_f16(v16h a, v16h b, v8f c) {
    // D = A(16x32 f16) * B(32x16 f16) + C(16x16 f32)
    return __builtin_amdgcn_wmma_f32_16x16x32_f16(
        false, a, false, b, (short)0, c, false, false);
}

// 16B global -> LDS copy; async on CDNA5 (ASYNCcnt), sync fallback otherwise
__device__ __forceinline__ void copy16_g2l(_Float16* ldst, const _Float16* gsrc) {
#if HAVE_ASYNC
    __builtin_amdgcn_global_load_async_to_lds_b128(
        (gv4i_t*)(uintptr_t)gsrc,
        (sv4i_t*)(unsigned)(uintptr_t)ldst, 0, 0);
#else
    *(v8h*)ldst = *(const v8h*)gsrc;
#endif
}

__device__ __forceinline__ void wait_async() {
#if HAVE_ASYNC
    __builtin_amdgcn_s_wait_asynccnt(0);
#endif
}

// reductions across the 16-lane group holding one matrix row (wave32)
__device__ __forceinline__ float rowmax16(float v) {
#pragma unroll
    for (int m = 1; m < 16; m <<= 1) v = fmaxf(v, __shfl_xor(v, m, 32));
    return v;
}
__device__ __forceinline__ float rowsum16(float v) {
#pragma unroll
    for (int m = 1; m < 16; m <<= 1) v += __shfl_xor(v, m, 32);
    return v;
}

// ---------- kernel 1: weights -> f16 workspace ----------

__global__ __launch_bounds__(256) void wconv_kernel(
    const float* __restrict__ Wq, const float* __restrict__ bq,
    const float* __restrict__ Wk, const float* __restrict__ bk,
    const float* __restrict__ Wv, const float* __restrict__ bv,
    _Float16* __restrict__ Wh, float* __restrict__ Bh)
{
    int i = blockIdx.x * 256 + threadIdx.x;          // 0 .. 384*256-1
    if (i < 64 * 256)            Wh[i] = (_Float16)Wq[i];
    else if (i < 128 * 256)      Wh[i] = (_Float16)Wk[i - 64 * 256];
    else if (i < 384 * 256)      Wh[i] = (_Float16)Wv[i - 128 * 256];
    if (i < 64)                  Bh[i] = bq[i];
    else if (i < 128)            Bh[i] = bk[i - 64];
    else if (i < 384)            Bh[i] = bv[i - 128];
}

// ---------- kernel 2: fused Q/K/V projection (WMMA GEMM) ----------
// Qh[b][n][64], Kh[b][n][64], Vh[b][c][n]   (all f16)

__global__ __launch_bounds__(256) void proj_kernel(
    const float* __restrict__ x, const _Float16* __restrict__ Wh,
    const float* __restrict__ Bh,
    _Float16* __restrict__ Qh, _Float16* __restrict__ Kh, _Float16* __restrict__ Vh)
{
    const int b    = blockIdx.y;
    const int n0   = blockIdx.x * 16;
    const int tid  = threadIdx.x;
    const int wave = tid >> 5;
    const int lane = tid & 31;
    const int l16  = lane & 15;
    const int hi   = lane >> 4;

    __shared__ _Float16 sX[16 * 256];                // x^T tile [n'][c]
    __shared__ _Float16 sT[8][16 * 16];              // per-wave transpose scratch

    // cooperative load + f32->f16 transpose: x[b][c][n0+n'] -> sX[n'][c]
    for (int idx = tid; idx < 16 * 256; idx += 256) {
        int c  = idx >> 4;
        int np = idx & 15;
        sX[np * 256 + c] = (_Float16)x[((size_t)b * CC + c) * NN + n0 + np];
    }
    __syncthreads();

    // A fragments over the full K=256 reduction (8 chunks of 32)
    v16h af[8];
#pragma unroll
    for (int kc = 0; kc < 8; ++kc) {
        const _Float16* p = sX + l16 * 256 + kc * 32;
        af[kc] = ld16_split(p + 8 * hi, p + 16 + 8 * hi);
    }

    // 24 output-column tiles (Q:0-3, K:4-7, V:8-23); 3 per wave
#pragma unroll
    for (int t = 0; t < 3; ++t) {
        const int ct = wave * 3 + t;
        const int o0 = ct * 16;
        v8f acc = {};
#pragma unroll
        for (int kc = 0; kc < 8; ++kc) {
            const _Float16* wp = Wh + (size_t)(o0 + l16) * 256 + kc * 32 + 8 * hi;
            acc = wmma_f16(af[kc], ld16_split(wp, wp + 16), acc);
        }
        const float bias = Bh[o0 + l16];

        if (o0 < 128) {
            // Q/K: lanes walk o (contiguous) -> direct stores are coalesced
            const int o = o0 + l16;
#pragma unroll
            for (int r = 0; r < 8; ++r) {
                const int n = n0 + r + 8 * hi;
                const _Float16 val = (_Float16)(acc[r] + bias);
                if (o0 < 64) Qh[((size_t)b * NN + n) * DD + o] = val;
                else         Kh[((size_t)b * NN + n) * DD + (o - 64)] = val;
            }
        } else {
            // V: transpose 16x16 tile through LDS so stores run along n
            _Float16* tb = sT[wave];
            __threadfence_block();                    // WAR vs previous t
#pragma unroll
            for (int r = 0; r < 8; ++r)
                tb[l16 * 16 + (r + 8 * hi)] = (_Float16)(acc[r] + bias);
            __threadfence_block();                    // RAW: stores -> reads
            const int row2 = lane >> 1;               // c' 0..15
            const int half = lane & 1;                // n half
            const int c = (o0 - 128) + row2;
            _Float16* dst = Vh + ((size_t)b * CC + c) * NN + n0 + half * 8;
            *(v8h*)dst = *(const v8h*)(tb + row2 * 16 + half * 8);
        }
    }
}

// ---------- kernel 3: flash attention + residual ----------
// wave -> (qtile = wave/2: 16 query rows, chalf = wave&1: 128 channels)

__global__ __launch_bounds__(256) void flash_kernel(
    const float* __restrict__ x,
    const _Float16* __restrict__ Qh, const _Float16* __restrict__ Kh,
    const _Float16* __restrict__ Vh, float* __restrict__ out)
{
    const int b     = blockIdx.y;
    const int tid   = threadIdx.x;
    const int wave  = tid >> 5;
    const int lane  = tid & 31;
    const int l16   = lane & 15;
    const int hi    = lane >> 4;
    const int qtile = wave >> 1;
    const int chalf = wave & 1;
    const int q0    = blockIdx.x * 64 + qtile * 16;

    __shared__ _Float16 sK[2][32 * 64];      // K^T chunk  [j'][d], double-buffered
    __shared__ _Float16 sV[2][256 * 32];     // V   chunk  [c][j'], double-buffered
    __shared__ _Float16 sP[8][16 * 32];      // per-wave P [i'][j']

    // Q tile as two A-fragments (d = 0..31, 32..63), kept in registers
    const _Float16* qrow = Qh + ((size_t)b * NN + q0) * DD;
    v16h qa[2];
#pragma unroll
    for (int kd = 0; kd < 2; ++kd) {
        const _Float16* p = qrow + (size_t)l16 * DD + kd * 32;
        qa[kd] = ld16_split(p + 8 * hi, p + 16 + 8 * hi);
    }

    v8f Oacc[8];
#pragma unroll
    for (int ct = 0; ct < 8; ++ct) Oacc[ct] = {};
    float m_[8], l_[8];
#pragma unroll
    for (int r = 0; r < 8; ++r) { m_[r] = -3.0e38f; l_[r] = 0.0f; }

    // cooperative async stage of one 32-key chunk into buffer `buf`
    auto stage = [&](int buf, int j0) {
        const int row = tid >> 3;                    // 0..31
        const int off = (tid & 7) * 8;               // 0..56 halves
        copy16_g2l(&sK[buf][row * 64 + off],
                   Kh + (size_t)(b * NN + j0 + row) * DD + off);
#pragma unroll
        for (int kk = 0; kk < 4; ++kk) {
            const int idx = tid + kk * 256;
            const int c   = idx >> 2;                // 0..255
            const int o2  = (idx & 3) * 8;           // 0..24 halves
            copy16_g2l(&sV[buf][c * 32 + o2],
                       Vh + ((size_t)b * CC + c) * NN + j0 + o2);
        }
    };

    stage(0, 0);
    wait_async();
    __syncthreads();

    for (int j0 = 0; j0 < NN; j0 += 32) {
        const int cur = (j0 >> 5) & 1;
        if (j0 + 32 < NN) stage(cur ^ 1, j0 + 32);   // overlap with compute

        // S = Q * K^T : two 16x16 subtiles, K-dim = 64 (2 WMMAs each)
        v8f S[2];
#pragma unroll
        for (int s = 0; s < 2; ++s) {
            v8f acc = {};
#pragma unroll
            for (int kd = 0; kd < 2; ++kd) {
                const _Float16* kp = &sK[cur][(s * 16 + l16) * 64 + kd * 32 + 8 * hi];
                acc = wmma_f16(qa[kd], ld16_split(kp, kp + 16), acc);
            }
            S[s] = acc;
        }

        // online softmax (rows r + 8*hi live in 16-lane groups)
        float alpha[8];
#pragma unroll
        for (int r = 0; r < 8; ++r) {
            float mx = rowmax16(fmaxf(S[0][r], S[1][r]));
            float mn = fmaxf(m_[r], mx);
            alpha[r] = __expf(m_[r] - mn);
            m_[r]    = mn;
            S[0][r]  = __expf(S[0][r] - mn);
            S[1][r]  = __expf(S[1][r] - mn);
            l_[r]    = l_[r] * alpha[r] + rowsum16(S[0][r] + S[1][r]);
        }
#pragma unroll
        for (int ct = 0; ct < 8; ++ct)
#pragma unroll
            for (int r = 0; r < 8; ++r)
                Oacc[ct][r] *= alpha[r];

        // D-layout P -> LDS -> A-layout fragment
        _Float16* myP = sP[wave];
#pragma unroll
        for (int s = 0; s < 2; ++s)
#pragma unroll
            for (int r = 0; r < 8; ++r)
                myP[(r + 8 * hi) * 32 + s * 16 + l16] = (_Float16)S[s][r];
        __syncthreads();

        const v16h pa = ld16_split(myP + l16 * 32 + 8 * hi,
                                   myP + l16 * 32 + 16 + 8 * hi);

        // O += P * V^T over this wave's 128 channels (8 c-tiles)
#pragma unroll
        for (int ct = 0; ct < 8; ++ct) {
            const _Float16* vp = &sV[cur][(chalf * 128 + ct * 16 + l16) * 32 + 8 * hi];
            Oacc[ct] = wmma_f16(pa, ld16_split(vp, vp + 16), Oacc[ct]);
        }

        wait_async();        // next chunk resident before it becomes `cur`
        __syncthreads();     // and everyone done reading `cur` before restage
    }

    // finalize: normalize, transpose tiles through LDS, coalesced store + residual
    float inv[8];
#pragma unroll
    for (int r = 0; r < 8; ++r) inv[r] = 1.0f / l_[r];

    float* pf = (float*)sP[wave];                    // 1 KB per wave, reuse
    const int row2 = lane >> 1;                      // c' 0..15
    const int half = lane & 1;                       // i half
#pragma unroll
    for (int ct = 0; ct < 8; ++ct) {
        __threadfence_block();                       // WAR vs previous ct reads
#pragma unroll
        for (int r = 0; r < 8; ++r)
            pf[l16 * 16 + (r + 8 * hi)] = Oacc[ct][r] * inv[r];
        __threadfence_block();                       // RAW: stores -> reads
        const int c = chalf * 128 + ct * 16 + row2;
        const size_t obase = ((size_t)b * CC + c) * NN + q0 + half * 8;
#pragma unroll
        for (int u = 0; u < 8; ++u) {
            const float v = pf[row2 * 16 + half * 8 + u];
            out[obase + u] = v + x[obase + u];       // contiguous 32B per lane
        }
    }
}

// ---------- launcher ----------

extern "C" void kernel_launch(void* const* d_in, const int* in_sizes, int n_in,
                              void* d_out, int out_size, void* d_ws, size_t ws_size,
                              hipStream_t stream)
{
    const float* x  = (const float*)d_in[0];
    const float* Wq = (const float*)d_in[1];
    const float* bq = (const float*)d_in[2];
    const float* Wk = (const float*)d_in[3];
    const float* bk = (const float*)d_in[4];
    const float* Wv = (const float*)d_in[5];
    const float* bv = (const float*)d_in[6];
    float* out = (float*)d_out;

    char* ws = (char*)d_ws;
    _Float16* Wh = (_Float16*)(ws);                         // 384*256*2   = 196608
    float*    Bh = (float*)(ws + 196608);                   // 384*4       -> 198144
    _Float16* Qh = (_Float16*)(ws + 198144);                // 2*8192*64*2 = 2097152
    _Float16* Kh = (_Float16*)(ws + 198144 + 2097152);
    _Float16* Vh = (_Float16*)(ws + 198144 + 2 * 2097152);  // 2*256*8192*2 = 8388608

    wconv_kernel<<<384, 256, 0, stream>>>(Wq, bq, Wk, bk, Wv, bv, Wh, Bh);
    proj_kernel<<<dim3(NN / 16, BB), 256, 0, stream>>>(x, Wh, Bh, Qh, Kh, Vh);
    flash_kernel<<<dim3(NN / 64, BB), 256, 0, stream>>>(x, Qh, Kh, Vh, out);
}
